// AttentionKANLayer_29996051595742
// MI455X (gfx1250) — matrix operand order repaired and checked by reading, not compiled, not run on hardware
//
#include <hip/hip_runtime.h>

// Problem constants (match reference)
#define N_FEAT   64
#define HID      10
#define OUT_D    5
#define BATCH_N  4096
#define N_PAIRS  2016      // 64*63/2
#define WPB      16        // waves (batch rows) per block
#define BLOCK_T  (WPB * 32)

typedef __attribute__((ext_vector_type(16))) _Float16 v16h;
typedef __attribute__((ext_vector_type(8)))  float    v8f;

__global__ __launch_bounds__(BLOCK_T) void attn_kan_fused(
    const float* __restrict__ x,
    const float* __restrict__ phi_W1, const float* __restrict__ phi_b1,
    const float* __restrict__ phi_W2, const float* __restrict__ phi_b2,
    const float* __restrict__ attn_W, const float* __restrict__ attn_b,
    const float* __restrict__ psi_W1, const float* __restrict__ psi_b1,
    const float* __restrict__ psi_W2, const float* __restrict__ psi_b2,
    float* __restrict__ out)
{
    // Per-wave b-projection rows (read back as uniform broadcasts), 40 KB.
    __shared__ float bProj[WPB][N_FEAT][HID];
    // 16 S-rows, zero-padded to K=32 so the WMMA A operand needs no guards. 2 KB.
    __shared__ float sTile[WPB][32];

    const int lane = threadIdx.x & 31;
    const int wid  = threadIdx.x >> 5;
    const int b    = blockIdx.x * WPB + wid;

    // Zero-pad this wave's S row (cols >= HID stay 0 for the WMMA A build).
    sTile[wid][lane] = 0.0f;

    // ---- phi + attention score: lane owns features i=lane and i=lane+32 ----
    float po[2][OUT_D];
    float sc[2];
#pragma unroll
    for (int f = 0; f < 2; ++f) {
        const int   i  = lane + f * 32;
        const float xi = x[b * N_FEAT + i];
        float h[HID];
#pragma unroll
        for (int k = 0; k < HID; ++k) {
            const float t = fmaf(xi, phi_W1[k], phi_b1[k]);
            h[k] = t > 0.0f ? t : 0.0f;
        }
#pragma unroll
        for (int o = 0; o < OUT_D; ++o) {
            float acc = phi_b2[o];
#pragma unroll
            for (int k = 0; k < HID; ++k) acc = fmaf(h[k], phi_W2[k * OUT_D + o], acc);
            po[f][o] = acc;
        }
        float s = attn_b[0];
#pragma unroll
        for (int o = 0; o < OUT_D; ++o) s = fmaf(po[f][o], attn_W[o], s);
        sc[f] = s;
    }

    // ---- softmax over the wave's 64 scores (wave32 butterfly reductions) ----
    float m = fmaxf(sc[0], sc[1]);
#pragma unroll
    for (int off = 16; off >= 1; off >>= 1) m = fmaxf(m, __shfl_xor(m, off, 32));
    const float e0 = __expf(sc[0] - m);
    const float e1 = __expf(sc[1] - m);
    float den = e0 + e1;
#pragma unroll
    for (int off = 16; off >= 1; off >>= 1) den += __shfl_xor(den, off, 32);
    const float inv = 1.0f / den;

    float wo[2][OUT_D];
#pragma unroll
    for (int o = 0; o < OUT_D; ++o) {
        wo[0][o] = po[0][o] * (e0 * inv);
        wo[1][o] = po[1][o] * (e1 * inv);
    }

    // ---- psi input projections: a (psi_W1 rows 0..4, + psi_b1) stays in
    //      registers; b (rows 5..9) goes to LDS for broadcast in the pair loop.
    float aP[2][HID];
#pragma unroll
    for (int f = 0; f < 2; ++f) {
#pragma unroll
        for (int k = 0; k < HID; ++k) {
            float acc = psi_b1[k];
#pragma unroll
            for (int o = 0; o < OUT_D; ++o)
                acc = fmaf(wo[f][o], psi_W1[o * HID + k], acc);
            aP[f][k] = acc;
        }
    }
#pragma unroll
    for (int f = 0; f < 2; ++f) {
#pragma unroll
        for (int k = 0; k < HID; ++k) {
            float acc = 0.0f;
#pragma unroll
            for (int o = 0; o < OUT_D; ++o)
                acc = fmaf(wo[f][o], psi_W1[(OUT_D + o) * HID + k], acc);
            bProj[wid][lane + f * 32][k] = acc;
        }
    }

    __syncthreads();

    // ---- pair loop: S[k] = sum_{i<j} relu(a_i[k] + b_j[k])  (bias folded in a)
    float S[HID];
#pragma unroll
    for (int k = 0; k < HID; ++k) S[k] = 0.0f;

    for (int j = 1; j < N_FEAT; ++j) {
        float bj[HID];
#pragma unroll
        for (int k = 0; k < HID; ++k) bj[k] = bProj[wid][j][k];  // LDS broadcast
        if (lane < j) {
#pragma unroll
            for (int k = 0; k < HID; ++k) S[k] += fmaxf(aP[0][k] + bj[k], 0.0f);
        }
        if (lane + 32 < j) {
#pragma unroll
            for (int k = 0; k < HID; ++k) S[k] += fmaxf(aP[1][k] + bj[k], 0.0f);
        }
    }

    // wave-reduce S, lane 0 publishes the row
#pragma unroll
    for (int k = 0; k < HID; ++k) {
#pragma unroll
        for (int off = 16; off >= 1; off >>= 1) S[k] += __shfl_xor(S[k], off, 32);
    }
    if (lane == 0) {
#pragma unroll
        for (int k = 0; k < HID; ++k) sTile[wid][k] = S[k];
    }
    __syncthreads();

    // ---- wave 0: out(16x5) = S(16x10) @ psi_W2(10x5) via one f16 WMMA ----
    if (threadIdx.x < 32) {
        const int hlf = lane >> 4;
        const int idx = lane & 15;   // row for A; col for B and C/D
        v16h Am, Bm;
        v8f  Cm = {};
#pragma unroll
        for (int e = 0; e < 16; ++e) {
            // 16-bit A layout: lanes 0-15 K = {0..7, 16..23}; lanes 16-31 K = {8..15, 24..31}
            const int kA = (e < 8) ? (hlf * 8 + e) : (16 + hlf * 8 + (e - 8));
            Am[e] = (_Float16)sTile[idx][kA];            // zero-padded beyond K=9
            // B layout: lanes 0-15 K = {0..7, 16..23}; lanes 16-31 K = {8..15, 24..31}; col = lane&15
            const int kB = kA;
            float bv = 0.0f;
            if (kB < HID && idx < OUT_D) bv = psi_W2[kB * OUT_D + idx];
            Bm[e] = (_Float16)bv;
        }
        v8f D = __builtin_amdgcn_wmma_f32_16x16x32_f16(
            /*neg_a=*/false, Am, /*neg_b=*/false, Bm,
            /*c_mod=*/(short)0, Cm, /*reuse_a=*/false, /*reuse_b=*/false);

        if (idx < OUT_D) {
            const float bias = (float)N_PAIRS * psi_b2[idx];
#pragma unroll
            for (int r = 0; r < 8; ++r) {
                const int row = r + hlf * 8;                  // C/D: VGPR r -> M=r / r+8
                const int bb  = blockIdx.x * WPB + row;
                out[bb * OUT_D + idx] = D[r] + bias;
            }
        }
    }
}

extern "C" void kernel_launch(void* const* d_in, const int* in_sizes, int n_in,
                              void* d_out, int out_size, void* d_ws, size_t ws_size,
                              hipStream_t stream) {
    (void)in_sizes; (void)n_in; (void)out_size; (void)d_ws; (void)ws_size;
    const float* X      = (const float*)d_in[0];
    const float* phi_W1 = (const float*)d_in[1];
    const float* phi_b1 = (const float*)d_in[2];
    const float* phi_W2 = (const float*)d_in[3];
    const float* phi_b2 = (const float*)d_in[4];
    const float* attn_W = (const float*)d_in[5];
    const float* attn_b = (const float*)d_in[6];
    const float* psi_W1 = (const float*)d_in[7];
    const float* psi_b1 = (const float*)d_in[8];
    const float* psi_W2 = (const float*)d_in[9];
    const float* psi_b2 = (const float*)d_in[10];
    float* out = (float*)d_out;

    dim3 grid(BATCH_N / WPB);
    dim3 block(BLOCK_T);
    attn_kan_fused<<<grid, block, 0, stream>>>(
        X, phi_W1, phi_b1, phi_W2, phi_b2, attn_W, attn_b,
        psi_W1, psi_b1, psi_W2, psi_b2, out);
}
